// EyetrackRegressionBiLSTM_11123965296980
// MI455X (gfx1250) — matrix operand order repaired and checked by reading, not compiled, not run on hardware
//
#include <hip/hip_runtime.h>
#include <hip/hip_bf16.h>
#include <cstdint>

#define IN_DIM 300
#define H_DIM  256
#define P_DIM  6
#define T_DIM  512
#define B_DIM  128
#define G_DIM  1024   // 4*H
#define BPW    8      // batch elements per workgroup (recurrence kernels)
#define NWG    (B_DIM / BPW)

typedef __bf16 bf16_t;
typedef bf16_t v16bf __attribute__((ext_vector_type(16)));
typedef float  v8f   __attribute__((ext_vector_type(8)));

static __device__ __forceinline__ float sigmoidf_(float x) {
  return 1.0f / (1.0f + __expf(-x));
}

// ---------------------------------------------------------------------------
// Kernel 1: xg0[m][g] = sum_k x[m][k]*Wih0[g][k] + bih0[g] + bhh0[g]
//   m = t*B + b,  x[m][k] = embedding[b][t][k]   (time-major view)
// One 16x16 output tile per wave via v_wmma_f32_16x16x32_bf16 (fp32 accum).
// ---------------------------------------------------------------------------
__global__ __launch_bounds__(256) void k_input_gemm(
    const float* __restrict__ emb, const float* __restrict__ Wih,
    const float* __restrict__ bih, const float* __restrict__ bhh,
    float* __restrict__ xg) {
  const int lane = threadIdx.x & 31;
  const int wave = threadIdx.x >> 5;
  const int m0 = (blockIdx.x * 8 + wave) * 16;   // row tile (t*B+b)
  const int n0 = blockIdx.y * 16;                // col tile (gate)
  const int l16 = lane & 15;
  const int hi  = lane >> 4;                     // lane-half select
  const int m = m0 + l16;                        // A row for this lane
  const int g = n0 + l16;                        // B column (gate) for this lane
  const int t = m >> 7;                          // /B_DIM
  const int b = m & (B_DIM - 1);
  const float* __restrict__ arow = emb + ((size_t)b * T_DIM + t) * IN_DIM;
  const float* __restrict__ brow = Wih + (size_t)g * IN_DIM;

  v8f acc = {};

  // 9 full K=32 chunks: max k = 287 < 300, no guards. 16B-aligned float4 loads.
#pragma unroll
  for (int kk = 0; kk < 288; kk += 32) {
    const int ka = kk + hi * 8;    // A: two 8-float runs at ka, ka+16
    const int kb = kk + hi * 8;    // B: two 8-float runs at kb, kb+16
    const float4 a0 = *(const float4*)(arow + ka);
    const float4 a1 = *(const float4*)(arow + ka + 4);
    const float4 a2 = *(const float4*)(arow + ka + 16);
    const float4 a3 = *(const float4*)(arow + ka + 20);
    const float4 b0 = *(const float4*)(brow + kb);
    const float4 b1 = *(const float4*)(brow + kb + 4);
    const float4 b2 = *(const float4*)(brow + kb + 16);
    const float4 b3 = *(const float4*)(brow + kb + 20);
    v16bf af, bfv;
    af[0]=(bf16_t)a0.x;  af[1]=(bf16_t)a0.y;  af[2]=(bf16_t)a0.z;  af[3]=(bf16_t)a0.w;
    af[4]=(bf16_t)a1.x;  af[5]=(bf16_t)a1.y;  af[6]=(bf16_t)a1.z;  af[7]=(bf16_t)a1.w;
    af[8]=(bf16_t)a2.x;  af[9]=(bf16_t)a2.y;  af[10]=(bf16_t)a2.z; af[11]=(bf16_t)a2.w;
    af[12]=(bf16_t)a3.x; af[13]=(bf16_t)a3.y; af[14]=(bf16_t)a3.z; af[15]=(bf16_t)a3.w;
    bfv[0]=(bf16_t)b0.x;  bfv[1]=(bf16_t)b0.y;  bfv[2]=(bf16_t)b0.z;  bfv[3]=(bf16_t)b0.w;
    bfv[4]=(bf16_t)b1.x;  bfv[5]=(bf16_t)b1.y;  bfv[6]=(bf16_t)b1.z;  bfv[7]=(bf16_t)b1.w;
    bfv[8]=(bf16_t)b2.x;  bfv[9]=(bf16_t)b2.y;  bfv[10]=(bf16_t)b2.z; bfv[11]=(bf16_t)b2.w;
    bfv[12]=(bf16_t)b3.x; bfv[13]=(bf16_t)b3.y; bfv[14]=(bf16_t)b3.z; bfv[15]=(bf16_t)b3.w;
    acc = __builtin_amdgcn_wmma_f32_16x16x32_bf16(
        false, af, false, bfv, (short)0, acc, false, false);
  }
  // Tail chunk kk=288 (k in [288,320)). All loads unconditional at clamped,
  // 16B-aligned in-bounds addresses; invalid elements zeroed via selects only.
  //   A lane-half hi: k = 288 + 8*hi + e (e<8 real; e>=8 always OOB -> 0)
  //   B lane-half hi: k = 288 + 8*hi + e (e<8 real; e>=8 always OOB -> 0)
  {
    const float4 t0 = *(const float4*)(arow + (hi ? 296 : 288)); // always valid ks
    const float4 t1 = *(const float4*)(arow + 292);              // valid when hi==0
    const float4 u0 = *(const float4*)(brow + (hi ? 296 : 288));
    const float4 u1 = *(const float4*)(brow + (hi ? 296 : 292));
    const float4 u2 = *(const float4*)(brow + 296);
    const float hz = hi ? 0.0f : 1.0f;
    v16bf af = {}, bfv = {};
    af[0]=(bf16_t)t0.x; af[1]=(bf16_t)t0.y; af[2]=(bf16_t)t0.z; af[3]=(bf16_t)t0.w;
    af[4]=(bf16_t)(hz*t1.x); af[5]=(bf16_t)(hz*t1.y);
    af[6]=(bf16_t)(hz*t1.z); af[7]=(bf16_t)(hz*t1.w);
    // af[8..15] stay 0
    bfv[0]=(bf16_t)u0.x;  bfv[1]=(bf16_t)u0.y;
    bfv[2]=(bf16_t)u0.z;  bfv[3]=(bf16_t)u0.w;
    bfv[4]=(bf16_t)(hz*u1.x);  bfv[5]=(bf16_t)(hz*u1.y);
    bfv[6]=(bf16_t)(hz*u1.z);  bfv[7]=(bf16_t)(hz*u1.w);
    // bfv[8..15] stay 0
    acc = __builtin_amdgcn_wmma_f32_16x16x32_bf16(
        false, af, false, bfv, (short)0, acc, false, false);
  }
  // C layout: VGPR r -> M = r + 8*hi, lane -> N = l16. Fold bias here.
  const float bias = bih[g] + bhh[g];
  const int mbase = m0 + hi * 8;
#pragma unroll
  for (int r = 0; r < 8; ++r)
    xg[(size_t)(mbase + r) * G_DIM + g] = acc[r] + bias;
}

// ---------------------------------------------------------------------------
// Kernel 2: layer-0 recurrence. One WG per 8 batch rows, persistent over T.
// LDS: Whh[1024][6], Whr[6][256], gates[8][1024], c[8][256], oc[8][256], h/hn.
// ---------------------------------------------------------------------------
__global__ __launch_bounds__(256) void k_lstm0(
    const float* __restrict__ xg, const float* __restrict__ Whh,
    const float* __restrict__ Whr, float* __restrict__ h1) {
  extern __shared__ float smem[];
  float* sWhh   = smem;                       // G*6
  float* sWhr   = sWhh + G_DIM * 6;           // 6*H
  float* sGates = sWhr + 6 * H_DIM;           // BPW*G
  float* sC     = sGates + BPW * G_DIM;       // BPW*H
  float* sOC    = sC + BPW * H_DIM;           // BPW*H
  float* sH     = sOC + BPW * H_DIM;          // BPW*8
  float* sHn    = sH + BPW * 8;               // BPW*8

  const int tid = threadIdx.x;
  const int b0 = blockIdx.x * BPW;

  for (int i = tid; i < G_DIM * 6; i += 256) sWhh[i] = Whh[i];
  for (int i = tid; i < 6 * H_DIM; i += 256) sWhr[i] = Whr[i];
  for (int i = tid; i < BPW * H_DIM; i += 256) sC[i] = 0.0f;
  if (tid < BPW * 8) sH[tid] = 0.0f;
  __syncthreads();

  for (int t = 0; t < T_DIM; ++t) {
    const float* __restrict__ xrow = xg + ((size_t)t * B_DIM + b0) * G_DIM;
    // Prefetch next step's 32KB gate slice (one 128B line per thread).
    if (t + 1 < T_DIM)
      __builtin_prefetch(xrow + (size_t)B_DIM * G_DIM + (size_t)tid * 32, 0, 0);
    // gates = xg_t + h @ Whh^T   (K = 6)
    for (int i = tid; i < BPW * G_DIM; i += 256) {
      const int bb = i >> 10;
      const int gg = i & (G_DIM - 1);
      float acc = xrow[i];
      const float* hb = sH + bb * 8;
      const float* w  = sWhh + gg * 6;
#pragma unroll
      for (int k = 0; k < 6; ++k) acc += hb[k] * w[k];
      sGates[i] = acc;
    }
    if (tid < BPW * 8) sHn[tid] = 0.0f;
    __syncthreads();
    // activations + cell update (i,f,g,o chunks of H)
    for (int i = tid; i < BPW * H_DIM; i += 256) {
      const int bb = i >> 8;
      const int j  = i & (H_DIM - 1);
      const float* gb = sGates + bb * G_DIM;
      const float ii = sigmoidf_(gb[j]);
      const float ff = sigmoidf_(gb[H_DIM + j]);
      const float gv = tanhf(gb[2 * H_DIM + j]);
      const float oo = sigmoidf_(gb[3 * H_DIM + j]);
      const float c  = ff * sC[i] + ii * gv;
      sC[i]  = c;
      sOC[i] = oo * tanhf(c);
    }
    __syncthreads();
    // projection: h_new[b][p] = oc[b][:] . Whr[p][:]  (32 threads/batch row)
    {
      const int bb = tid >> 5;
      const int jb = (tid & 31) * 8;
      float p0=0,p1=0,p2=0,p3=0,p4=0,p5=0;
#pragma unroll
      for (int q = 0; q < 8; ++q) {
        const float v = sOC[bb * H_DIM + jb + q];
        p0 += v * sWhr[0 * H_DIM + jb + q];
        p1 += v * sWhr[1 * H_DIM + jb + q];
        p2 += v * sWhr[2 * H_DIM + jb + q];
        p3 += v * sWhr[3 * H_DIM + jb + q];
        p4 += v * sWhr[4 * H_DIM + jb + q];
        p5 += v * sWhr[5 * H_DIM + jb + q];
      }
      atomicAdd(&sHn[bb * 8 + 0], p0);
      atomicAdd(&sHn[bb * 8 + 1], p1);
      atomicAdd(&sHn[bb * 8 + 2], p2);
      atomicAdd(&sHn[bb * 8 + 3], p3);
      atomicAdd(&sHn[bb * 8 + 4], p4);
      atomicAdd(&sHn[bb * 8 + 5], p5);
    }
    __syncthreads();
    if (tid < BPW * 8) {
      const int bb = tid >> 3, p = tid & 7;
      const float v = (p < P_DIM) ? sHn[tid] : 0.0f;
      sH[tid] = v;
      if (p < P_DIM)
        h1[((size_t)t * B_DIM + b0 + bb) * P_DIM + p] = v;
    }
    __syncthreads();
  }
}

// ---------------------------------------------------------------------------
// Kernel 3: layer-1 recurrence (K=6 input GEMM fused) + mask + head + lengths.
// ---------------------------------------------------------------------------
__global__ __launch_bounds__(256) void k_lstm1(
    const float* __restrict__ h1, const float* __restrict__ Wih,
    const float* __restrict__ bih, const float* __restrict__ bhh,
    const float* __restrict__ Whh, const float* __restrict__ Whr,
    const float* __restrict__ mask, const float* __restrict__ feature,
    const float* __restrict__ Wl, const float* __restrict__ bl,
    const int* __restrict__ lengths_w, float* __restrict__ out) {
  extern __shared__ float smem[];
  float* sWih   = smem;                       // G*6
  float* sWhh   = sWih + G_DIM * 6;           // G*6
  float* sBias  = sWhh + G_DIM * 6;           // G
  float* sWhr   = sBias + G_DIM;              // 6*H
  float* sGates = sWhr + 6 * H_DIM;           // BPW*G
  float* sC     = sGates + BPW * G_DIM;       // BPW*H
  float* sOC    = sC + BPW * H_DIM;           // BPW*H
  float* sX     = sOC + BPW * H_DIM;          // 64
  float* sH     = sX + 64;                    // 64
  float* sHn    = sH + 64;                    // 64
  float* sRO    = sHn + 64;                   // 64
  float* sWl    = sRO + 64;                   // 16

  const int tid = threadIdx.x;
  const int b0 = blockIdx.x * BPW;

  for (int i = tid; i < G_DIM * 6; i += 256) { sWih[i] = Wih[i]; sWhh[i] = Whh[i]; }
  for (int i = tid; i < G_DIM; i += 256) sBias[i] = bih[i] + bhh[i];
  for (int i = tid; i < 6 * H_DIM; i += 256) sWhr[i] = Whr[i];
  for (int i = tid; i < BPW * H_DIM; i += 256) sC[i] = 0.0f;
  if (tid < BPW * 8) sH[tid] = 0.0f;
  if (tid < 8) sWl[tid] = Wl[tid];
  if (tid == 8) sWl[8] = bl[0];
  // lengths passthrough (int64 bits) at tail of output
  if (blockIdx.x == 0 && tid < 2 * B_DIM)
    ((int*)(out + (size_t)T_DIM * B_DIM * 7))[tid] = lengths_w[tid];
  __syncthreads();

  for (int t = 0; t < T_DIM; ++t) {
    if (tid < BPW * P_DIM)
      sX[tid] = h1[((size_t)t * B_DIM + b0) * P_DIM + tid];
    if (tid < BPW * 8) sHn[tid] = 0.0f;
    __syncthreads();
    // gates = bias + x_t @ Wih^T + h @ Whh^T   (both K = 6)
    for (int i = tid; i < BPW * G_DIM; i += 256) {
      const int bb = i >> 10;
      const int gg = i & (G_DIM - 1);
      float acc = sBias[gg];
      const float* xb = sX + bb * P_DIM;
      const float* hb = sH + bb * 8;
      const float* wi = sWih + gg * 6;
      const float* wh = sWhh + gg * 6;
#pragma unroll
      for (int k = 0; k < 6; ++k) acc += xb[k] * wi[k] + hb[k] * wh[k];
      sGates[i] = acc;
    }
    __syncthreads();
    for (int i = tid; i < BPW * H_DIM; i += 256) {
      const int bb = i >> 8;
      const int j  = i & (H_DIM - 1);
      const float* gb = sGates + bb * G_DIM;
      const float ii = sigmoidf_(gb[j]);
      const float ff = sigmoidf_(gb[H_DIM + j]);
      const float gv = tanhf(gb[2 * H_DIM + j]);
      const float oo = sigmoidf_(gb[3 * H_DIM + j]);
      const float c  = ff * sC[i] + ii * gv;
      sC[i]  = c;
      sOC[i] = oo * tanhf(c);
    }
    __syncthreads();
    {
      const int bb = tid >> 5;
      const int jb = (tid & 31) * 8;
      float p0=0,p1=0,p2=0,p3=0,p4=0,p5=0;
#pragma unroll
      for (int q = 0; q < 8; ++q) {
        const float v = sOC[bb * H_DIM + jb + q];
        p0 += v * sWhr[0 * H_DIM + jb + q];
        p1 += v * sWhr[1 * H_DIM + jb + q];
        p2 += v * sWhr[2 * H_DIM + jb + q];
        p3 += v * sWhr[3 * H_DIM + jb + q];
        p4 += v * sWhr[4 * H_DIM + jb + q];
        p5 += v * sWhr[5 * H_DIM + jb + q];
      }
      atomicAdd(&sHn[bb * 8 + 0], p0);
      atomicAdd(&sHn[bb * 8 + 1], p1);
      atomicAdd(&sHn[bb * 8 + 2], p2);
      atomicAdd(&sHn[bb * 8 + 3], p3);
      atomicAdd(&sHn[bb * 8 + 4], p4);
      atomicAdd(&sHn[bb * 8 + 5], p5);
    }
    __syncthreads();
    // commit h2, masked return_output, stash for head
    if (tid < BPW * 8) {
      const int bb = tid >> 3, p = tid & 7;
      const float v = (p < P_DIM) ? sHn[tid] : 0.0f;
      sH[tid] = v;
      if (p < P_DIM) {
        const size_t tb = (size_t)t * B_DIM + b0 + bb;
        const float ro = v * mask[tb * 8 + 2 + p];
        out[(size_t)T_DIM * B_DIM + tb * P_DIM + p] = ro;
        sRO[tid] = ro;
      }
    }
    __syncthreads();
    // head: relu([feat0, feat1, ro0..ro5] . Wl + bl)
    if (tid < BPW) {
      const size_t tb = (size_t)t * B_DIM + b0 + tid;
      float acc = sWl[8] + feature[tb * 2 + 0] * sWl[0]
                         + feature[tb * 2 + 1] * sWl[1];
#pragma unroll
      for (int p = 0; p < P_DIM; ++p) acc += sRO[tid * 8 + p] * sWl[2 + p];
      out[tb] = fmaxf(acc, 0.0f);
    }
    __syncthreads();
  }
}

// ---------------------------------------------------------------------------
extern "C" void kernel_launch(void* const* d_in, const int* in_sizes, int n_in,
                              void* d_out, int out_size, void* d_ws, size_t ws_size,
                              hipStream_t stream) {
  (void)in_sizes; (void)n_in; (void)out_size; (void)ws_size;
  const float* emb     = (const float*)d_in[0];
  const float* mask    = (const float*)d_in[1];
  const float* feature = (const float*)d_in[2];
  const int*   lengths = (const int*)d_in[3];
  // d_in[4] = data_idx (unused)
  const float* Wih0 = (const float*)d_in[5];
  const float* Whh0 = (const float*)d_in[6];
  const float* bih0 = (const float*)d_in[7];
  const float* bhh0 = (const float*)d_in[8];
  const float* Whr0 = (const float*)d_in[9];
  const float* Wih1 = (const float*)d_in[10];
  const float* Whh1 = (const float*)d_in[11];
  const float* bih1 = (const float*)d_in[12];
  const float* bhh1 = (const float*)d_in[13];
  const float* Whr1 = (const float*)d_in[14];
  const float* Wl   = (const float*)d_in[15];
  const float* bl   = (const float*)d_in[16];
  float* out = (float*)d_out;

  float* xg0   = (float*)d_ws;                              // [T*B, G] fp32
  float* h1buf = xg0 + (size_t)T_DIM * B_DIM * G_DIM;       // [T*B, P] fp32

  dim3 blk(256);
  dim3 g1((T_DIM * B_DIM) / (8 * 16), G_DIM / 16);          // 512 x 64
  k_input_gemm<<<g1, blk, 0, stream>>>(emb, Wih0, bih0, bhh0, xg0);

  const size_t sh2 = (size_t)(G_DIM * 6 + 6 * H_DIM + BPW * G_DIM
                              + 2 * BPW * H_DIM + 2 * BPW * 8) * sizeof(float);
  (void)hipFuncSetAttribute((const void*)k_lstm0,
      hipFuncAttributeMaxDynamicSharedMemorySize, (int)sh2);
  k_lstm0<<<NWG, blk, sh2, stream>>>(xg0, Whh0, Whr0, h1buf);

  const size_t sh3 = (size_t)(2 * G_DIM * 6 + G_DIM + 6 * H_DIM + BPW * G_DIM
                              + 2 * BPW * H_DIM + 4 * 64 + 16) * sizeof(float);
  (void)hipFuncSetAttribute((const void*)k_lstm1,
      hipFuncAttributeMaxDynamicSharedMemorySize, (int)sh3);
  k_lstm1<<<NWG, blk, sh3, stream>>>(h1buf, Wih1, bih1, bhh1, Whh1, Whr1,
                                     mask, feature, Wl, bl, lengths, out);
}
